// Timeception_27419071217984
// MI455X (gfx1250) — matrix operand order repaired and checked by reading, not compiled, not run on hardware
//
#include <hip/hip_runtime.h>

// ---------------- types ----------------
typedef __attribute__((ext_vector_type(4)))  __bf16 bf16x4;
typedef __attribute__((ext_vector_type(8)))  __bf16 bf16x8;
typedef __attribute__((ext_vector_type(16))) __bf16 bf16x16;
typedef __attribute__((ext_vector_type(8)))  float  f32x8;

#define GRP 8

__device__ __forceinline__ __bf16 f2bf(float f) {
    unsigned u = __builtin_bit_cast(unsigned, f);
    u += 0x7FFFu + ((u >> 16) & 1u);           // round-to-nearest-even
    unsigned short h = (unsigned short)(u >> 16);
    return __builtin_bit_cast(__bf16, h);
}

// ---------------- prep: pack weights (bf16, B^T layout), bias, fold BN ----------------
__global__ void tc_prep_kernel(const float* __restrict__ W, const float* __restrict__ b,
                               const float* __restrict__ bn,
                               __bf16* __restrict__ WBt, float* __restrict__ biasP,
                               float* __restrict__ scaleP, float* __restrict__ shiftP,
                               int Cg, int KP, int cb, int N, int NP) {
    long tid = (long)blockIdx.x * blockDim.x + threadIdx.x;
    long nW = (long)GRP * NP * KP;
    if (tid < nW) {
        int k = (int)(tid % KP);
        long r = tid / KP;
        int n = (int)(r % NP);
        int g = (int)(r / NP);
        __bf16 out = __builtin_bit_cast(__bf16, (unsigned short)0);
        if (n < N && k < Cg) {
            int v = n / cb, o = n - v * cb;
            // W layout: (5, G, Cg, cb)
            out = f2bf(W[(((long)v * GRP + g) * Cg + k) * cb + o]);
        }
        WBt[tid] = out;   // layout [g][n][k]
        return;
    }
    tid -= nW;
    long nB = (long)GRP * NP;
    if (tid < nB) {
        int n = (int)(tid % NP);
        int g = (int)(tid / NP);
        float val = 0.f;
        if (n < N) {
            int v = n / cb, o = n - v * cb;
            val = b[((long)v * GRP + g) * cb + o];   // b layout (5,G,cb)
        }
        biasP[tid] = val;
        return;
    }
    tid -= nB;
    long nS = 5L * GRP * cb;
    if (tid < nS) {
        int o = (int)(tid % cb);
        long r = tid / cb;
        int g = (int)(r % GRP);
        int v = (int)(r / GRP);
        // bn layout (5,4,G,cb): [v][j][g][o]
        float gamma = bn[(((long)v * 4 + 0) * GRP + g) * cb + o];
        float beta  = bn[(((long)v * 4 + 1) * GRP + g) * cb + o];
        float mean  = bn[(((long)v * 4 + 2) * GRP + g) * cb + o];
        float var   = bn[(((long)v * 4 + 3) * GRP + g) * cb + o];
        float s = gamma * rsqrtf(var + 1e-3f);
        scaleP[tid] = s;              // index = (v*8+g)*cb+o
        shiftP[tid] = beta - mean * s;
    }
}

// ---------------- grouped pointwise-conv GEMM via bf16 WMMA ----------------
// grid: (M/128, NP/64, 8 groups), block: 256 threads (8 waves of 32)
#define LDP 40   // padded LDS row stride (elements) -> 80B rows, conflict-free frags
__global__ __launch_bounds__(256)
void tc_gemm_kernel(const float* __restrict__ X, const __bf16* __restrict__ WBt,
                    const float* __restrict__ biasP, float* __restrict__ Y,
                    int M, int Cin, int Cg, int KP, int N, int NP, int C5) {
    __shared__ __align__(16) __bf16 lds_a[128 * LDP];
    __shared__ __align__(16) __bf16 lds_b[64 * LDP];

    const int tid  = threadIdx.x;
    const int lane = tid & 31;
    const int wave = tid >> 5;
    const int wm   = wave >> 1;     // 0..3  (M tiles of 32)
    const int wn   = wave & 1;      // 0..1  (N tiles of 32)
    const int g    = blockIdx.z;
    const int mblk = blockIdx.x * 128;
    const int nblk = blockIdx.y * 64;

    const float*  Xg = X + (size_t)mblk * Cin + (size_t)g * Cg;
    const __bf16* Bg = WBt + ((size_t)g * NP + nblk) * KP;

    f32x8 acc[2][2];
#pragma unroll
    for (int i = 0; i < 2; ++i)
#pragma unroll
        for (int j = 0; j < 2; ++j)
#pragma unroll
            for (int e = 0; e < 8; ++e) acc[i][j][e] = 0.f;

    const int cLane  = lane & 15;
    const int hiHalf = lane >> 4;          // 0 or 1
    const int kA0    = hiHalf * 8;         // A frag: K base {0,8}
    const int kB0    = hiHalf * 8;         // B frag: K base {0,8}

    for (int k0 = 0; k0 < KP; k0 += 32) {
        // ---- stage A tile (128 rows x 32 k), fp32 -> bf16, with K zero-pad ----
#pragma unroll
        for (int j = 0; j < 4; ++j) {
            int idx = tid + j * 256;       // 0..1023
            int row = idx >> 3;            // 0..127
            int kk  = (idx & 7) * 4;       // 0,4,..,28
            int kg  = k0 + kk;
            const float* src = Xg + (size_t)row * Cin + kg;
            float f0 = (kg + 0 < Cg) ? src[0] : 0.f;
            float f1 = (kg + 1 < Cg) ? src[1] : 0.f;
            float f2 = (kg + 2 < Cg) ? src[2] : 0.f;
            float f3 = (kg + 3 < Cg) ? src[3] : 0.f;
            bf16x4 h;
            h[0] = f2bf(f0); h[1] = f2bf(f1); h[2] = f2bf(f2); h[3] = f2bf(f3);
            *(bf16x4*)&lds_a[row * LDP + kk] = h;
        }
        // ---- stage B tile (64 n-rows x 32 k), already bf16, zero-padded ----
        {
            int row = tid >> 2;            // 0..63
            int kk  = (tid & 3) * 8;       // 0,8,16,24
            bf16x8 h = *(const bf16x8*)(Bg + (size_t)row * KP + k0 + kk);
            *(bf16x8*)&lds_b[row * LDP + kk] = h;
        }
        // prefetch next A panel (gfx1250 global_prefetch_b8)
        if (k0 + 32 < KP) {
            int row = tid >> 1;
            int kk  = (tid & 1) * 16;
            __builtin_prefetch(Xg + (size_t)row * Cin + k0 + 32 + kk, 0, 1);
        }
        __syncthreads();

        // ---- fragments ----
        bf16x16 afrag[2], bfrag[2];
#pragma unroll
        for (int mi = 0; mi < 2; ++mi) {
            const __bf16* p = &lds_a[(wm * 32 + mi * 16 + cLane) * LDP + kA0];
            bf16x8 lo = *(const bf16x8*)p;          // K = kA0 .. kA0+7
            bf16x8 hi = *(const bf16x8*)(p + 16);   // K = kA0+16 .. kA0+23
#pragma unroll
            for (int e = 0; e < 8; ++e) { afrag[mi][e] = lo[e]; afrag[mi][e + 8] = hi[e]; }
        }
#pragma unroll
        for (int ni = 0; ni < 2; ++ni) {
            const __bf16* p = &lds_b[(wn * 32 + ni * 16 + cLane) * LDP + kB0];
            bf16x8 lo = *(const bf16x8*)p;          // K = kB0 .. kB0+7
            bf16x8 hi = *(const bf16x8*)(p + 16);   // K = kB0+16 .. kB0+23
#pragma unroll
            for (int e = 0; e < 8; ++e) { bfrag[ni][e] = lo[e]; bfrag[ni][e + 8] = hi[e]; }
        }
        // ---- 4 WMMAs per wave ----
#pragma unroll
        for (int mi = 0; mi < 2; ++mi)
#pragma unroll
            for (int ni = 0; ni < 2; ++ni)
                acc[mi][ni] = __builtin_amdgcn_wmma_f32_16x16x32_bf16(
                    false, afrag[mi], false, bfrag[ni],
                    (short)0, acc[mi][ni], false, false);
        __syncthreads();
    }

    // ---- bias + store (guard ragged N) ----
    const int rOff = hiHalf * 8;
#pragma unroll
    for (int mi = 0; mi < 2; ++mi)
#pragma unroll
        for (int ni = 0; ni < 2; ++ni) {
            int ncol = nblk + wn * 32 + ni * 16 + cLane;
            if (ncol < N) {
                float bias = biasP[g * NP + ncol];
                size_t col = (size_t)g * N + ncol;
#pragma unroll
                for (int r = 0; r < 8; ++r) {
                    int mrow = mblk + wm * 32 + mi * 16 + r + rOff;
                    Y[(size_t)mrow * C5 + col] = acc[mi][ni][r] + bias;
                }
            }
        }
}

// ---------------- fused epilogue: dwconv-t / maxpool / BN / relu / shuffle / t-pool ----------------
__global__ void tc_post_kernel(const float* __restrict__ Y, const float* __restrict__ dw,
                               const float* __restrict__ scaleP, const float* __restrict__ shiftP,
                               float* __restrict__ Xout,
                               int Tin, int cb, int N, int C5, long count) {
    long idx = (long)blockIdx.x * blockDim.x + threadIdx.x;
    if (idx >= count) return;
    int c = (int)(idx % C5);
    long r = idx / C5;
    int w = (int)(r % 7); r /= 7;
    int h = (int)(r % 7); r /= 7;
    int Tout = Tin >> 1;
    int t2 = (int)(r % Tout);
    int bb = (int)(r / Tout);
    // channel shuffle inverse: c_out = (v*cb+o)*8 + g
    int g = c & 7;
    int p = c >> 3;
    int v = p / cb;
    int o = p - v * cb;
    long ycol = (long)g * N + p;          // g*N + v*cb + o
    int hw = h * 7 + w;
    long rowBase = (long)bb * Tin * 49 + hw;      // + t*49 per timestep
    long tStride = 49L * C5;
    int si = (v * 8 + g) * cb + o;
    float s  = scaleP[si];
    float sh = shiftP[si];
    float vmax = 0.f;                      // relu outputs >= 0
    int t0 = t2 * 2;
#pragma unroll
    for (int dt = 0; dt < 2; ++dt) {
        int t = t0 + dt;
        long rt = (rowBase + (long)t * 49) * C5 + ycol;
        float val;
        if (v == 0) {
            val = Y[rt];
        } else if (v <= 3) {
            int d = v;
            float a1 = Y[rt];
            float a0 = (t - d >= 0)  ? Y[rt - (long)d * tStride] : 0.f;
            float a2 = (t + d < Tin) ? Y[rt + (long)d * tStride] : 0.f;
            // dw layout (3,3,G,cb): ((i*3+tap)*8+g)*cb+o, i = v-1
            long base = (long)(v - 1) * 3;
            float w0 = dw[((base + 0) * 8 + g) * cb + o];
            float w1 = dw[((base + 1) * 8 + g) * cb + o];
            float w2 = dw[((base + 2) * 8 + g) * cb + o];
            val = w0 * a0 + w1 * a1 + w2 * a2;
        } else {
            int t1 = (t + 1 < Tin) ? t + 1 : Tin - 1;
            float a = Y[rt];
            float b2 = Y[(rowBase + (long)t1 * 49) * C5 + ycol];
            val = fmaxf(a, b2);
        }
        val = fmaxf(val * s + sh, 0.f);    // BN + ReLU
        vmax = fmaxf(vmax, val);           // temporal maxpool (2,1,1)
    }
    Xout[idx] = vmax;
}

// ---------------- host ----------------
extern "C" void kernel_launch(void* const* d_in, const int* in_sizes, int n_in,
                              void* d_out, int out_size, void* d_ws, size_t ws_size,
                              hipStream_t stream) {
    (void)in_sizes; (void)n_in; (void)out_size; (void)ws_size;
    const float* x0 = (const float*)d_in[0];

    // workspace layout (256B-aligned)
    char* ws = (char*)d_ws;
    const long OFF_Y   = 0;                     // 256,901,120 B (max y: 50176*1280 f32)
    const long OFF_XA  = OFF_Y  + 256901120L;   // 128,450,560 B
    const long OFF_XB  = OFF_XA + 128450560L;   //  80,281,600 B
    const long OFF_WB  = OFF_XB + 80281600L;    //   1,310,720 B (max packed W bf16)
    const long OFF_BI  = OFF_WB + 1310720L;     //      10,240 B
    const long OFF_SC  = OFF_BI + 10240L;       //      10,240 B
    const long OFF_SH  = OFF_SC + 10240L;       //      10,240 B
    float*  Ybuf   = (float*)(ws + OFF_Y);
    float*  XA     = (float*)(ws + OFF_XA);
    float*  XB     = (float*)(ws + OFF_XB);
    __bf16* WBt    = (__bf16*)(ws + OFF_WB);
    float*  biasP  = (float*)(ws + OFF_BI);
    float*  scaleP = (float*)(ws + OFF_SC);
    float*  shiftP = (float*)(ws + OFF_SH);

    const int TinA[4] = {64, 32, 16, 8};
    const int CinA[4] = {1024, 1280, 1600, 2000};
    const int cbA[4]  = {32, 40, 50, 62};
    float* outs[4] = {XA, XB, XA, (float*)d_out};

    const float* cur = x0;
    for (int l = 0; l < 4; ++l) {
        const float* W  = (const float*)d_in[1 + 4 * l];
        const float* b  = (const float*)d_in[2 + 4 * l];
        const float* dw = (const float*)d_in[3 + 4 * l];
        const float* bn = (const float*)d_in[4 + 4 * l];
        int Tin = TinA[l], Cin = CinA[l], cb = cbA[l];
        int Cg = Cin / 8;
        int KP = (Cg + 31) & ~31;
        int N  = 5 * cb;
        int NP = (N + 63) & ~63;
        int M  = 16 * Tin * 49;
        int C5 = 8 * N;

        long prepTotal = (long)8 * NP * KP + 8L * NP + 5L * 8 * cb;
        tc_prep_kernel<<<dim3((unsigned)((prepTotal + 255) / 256)), dim3(256), 0, stream>>>(
            W, b, bn, WBt, biasP, scaleP, shiftP, Cg, KP, cb, N, NP);

        dim3 gg((unsigned)(M / 128), (unsigned)(NP / 64), 8);
        tc_gemm_kernel<<<gg, dim3(256), 0, stream>>>(
            cur, WBt, biasP, Ybuf, M, Cin, Cg, KP, N, NP, C5);

        long cnt = (long)16 * (Tin / 2) * 49 * C5;
        tc_post_kernel<<<dim3((unsigned)((cnt + 255) / 256)), dim3(256), 0, stream>>>(
            Ybuf, dw, scaleP, shiftP, outs[l], Tin, cb, N, C5, cnt);

        cur = outs[l];
    }
}
